// Model_65317862637833
// MI455X (gfx1250) — matrix operand order repaired and checked by reading, not compiled, not run on hardware
//
#include <hip/hip_runtime.h>
#include <math.h>

#define BB   32
#define LL   512
#define CIN  64
#define DD   512
#define DFFC 2048
#define BLTOK (BB * LL)      // 16384
#define TOPK 6

typedef __attribute__((ext_vector_type(16))) __bf16 bf16x16;
typedef __attribute__((ext_vector_type(8)))  float  v8f;

struct __align__(16) Q16 { unsigned int a, b, c, d; };
struct __align__(32) Q32 { Q16 lo, hi; };

__device__ __forceinline__ __bf16 f2bf(float f) {
  unsigned int u = __builtin_bit_cast(unsigned int, f);
  unsigned short s = (unsigned short)((u + 0x7fffu + ((u >> 16) & 1u)) >> 16);
  return __builtin_bit_cast(__bf16, s);
}

// ---------------------------------------------------------------------------
// WMMA GEMM: C[M,N] = A[M,K](bf16,row) * B[N,K](bf16,row, i.e. B^T of [K,N]) (+bias)(+relu)(+resid)
// 128x128 tile, 8 waves (4Mx2N), wave = 32x64 = 2x4 frags of 16x16, K-step 32.
// ---------------------------------------------------------------------------
__global__ void __launch_bounds__(256)
gemm_bf16_k(const __bf16* __restrict__ A, const __bf16* __restrict__ Bm,
            float* __restrict__ C, __bf16* __restrict__ Cbf,
            const float* __restrict__ bias, const float* __restrict__ resid,
            int M, int N, int K,
            long long sA, long long sB, long long sC, int relu)
{
  __shared__ __align__(16) __bf16 lA[128 * 40];
  __shared__ __align__(16) __bf16 lB[128 * 40];

  const int t    = threadIdx.x;
  const int lane = t & 31;
  const int wave = t >> 5;
  const int half = lane >> 4;      // 0: lanes 0-15, 1: lanes 16-31
  const int l15  = lane & 15;
  const int wm   = (wave & 3) * 32;    // wave tile origin in M (within block tile)
  const int wn   = (wave >> 2) * 64;   // wave tile origin in N

  const long long z = blockIdx.z;
  const __bf16* Ap = A  + (size_t)z * (size_t)sA;
  const __bf16* Bp = Bm + (size_t)z * (size_t)sB;
  const size_t  cz = (size_t)z * (size_t)sC;

  const int mBase = blockIdx.y * 128;
  const int nBase = blockIdx.x * 128;

  v8f acc[2][4];
#pragma unroll
  for (int i = 0; i < 2; ++i)
#pragma unroll
    for (int j = 0; j < 4; ++j)
#pragma unroll
      for (int e = 0; e < 8; ++e) acc[i][j][e] = 0.0f;

  for (int k0 = 0; k0 < K; k0 += 32) {
    __syncthreads();
    // Stage 128x32 A-tile and 128x32 B-tile (both K-contiguous) into LDS (pitch 40).
#pragma unroll
    for (int i = 0; i < 2; ++i) {
      int c   = t + 256 * i;        // 512 chunks of 8 bf16
      int row = c >> 2;
      int col = (c & 3) << 3;
      *reinterpret_cast<Q16*>(&lA[row * 40 + col]) =
          *reinterpret_cast<const Q16*>(Ap + (size_t)(mBase + row) * K + k0 + col);
      *reinterpret_cast<Q16*>(&lB[row * 40 + col]) =
          *reinterpret_cast<const Q16*>(Bp + (size_t)(nBase + row) * K + k0 + col);
    }
    __syncthreads();

    bf16x16 afr[2], bfr[4];
#pragma unroll
    for (int fm = 0; fm < 2; ++fm) {
      // 16-bit A layout: lanes0-15 rows hold K {0..7,16..23}; lanes16-31 hold K {8..15,24..31}
      int r  = wm + fm * 16 + l15;
      int c0 = half * 8;
      Q32 tmp;
      tmp.lo = *reinterpret_cast<const Q16*>(&lA[r * 40 + c0]);
      tmp.hi = *reinterpret_cast<const Q16*>(&lA[r * 40 + c0 + 16]);
      afr[fm] = __builtin_bit_cast(bf16x16, tmp);
    }
#pragma unroll
    for (int fn = 0; fn < 4; ++fn) {
      // 16-bit B layout: lanes0-15 (col n) hold K {0..7,16..23}; lanes16-31 hold K {8..15,24..31}
      int r  = wn + fn * 16 + l15;
      int c0 = half * 8;
      Q32 tmp;
      tmp.lo = *reinterpret_cast<const Q16*>(&lB[r * 40 + c0]);
      tmp.hi = *reinterpret_cast<const Q16*>(&lB[r * 40 + c0 + 16]);
      bfr[fn] = __builtin_bit_cast(bf16x16, tmp);
    }
#pragma unroll
    for (int fm = 0; fm < 2; ++fm)
#pragma unroll
      for (int fn = 0; fn < 4; ++fn)
        acc[fm][fn] = __builtin_amdgcn_wmma_f32_16x16x32_bf16(
            false, afr[fm], false, bfr[fn], (short)0, acc[fm][fn], false, false);
  }

  // Epilogue. C/D layout: VGPR j holds M=j (lanes 0-15) / M=j+8 (lanes 16-31), N=lane&15.
#pragma unroll
  for (int fm = 0; fm < 2; ++fm) {
#pragma unroll
    for (int fn = 0; fn < 4; ++fn) {
      int gcol = nBase + wn + fn * 16 + l15;
      float bv = bias ? bias[gcol] : 0.0f;
#pragma unroll
      for (int j = 0; j < 8; ++j) {
        int grow = mBase + wm + fm * 16 + half * 8 + j;
        float val = acc[fm][fn][j] + bv;
        if (relu) val = fmaxf(val, 0.0f);
        size_t idx = cz + (size_t)grow * (size_t)N + (size_t)gcol;
        if (resid) val += resid[idx];
        if (Cbf) Cbf[idx] = f2bf(val);
        else     C[idx]   = val;
      }
    }
  }
}

// ---------------------------------------------------------------------------
// Elementwise / helper kernels
// ---------------------------------------------------------------------------
__global__ void cvt_f32_bf16_k(const float* __restrict__ in, __bf16* __restrict__ out, int n) {
  int i = blockIdx.x * blockDim.x + threadIdx.x;
  if (i < n) out[i] = f2bf(in[i]);
}

// out[n*K + k] = (bf16) W[k*N + n]   (transpose + convert)
__global__ void wtrans_k(const float* __restrict__ W, __bf16* __restrict__ out, int K, int N) {
  int i = blockIdx.x * blockDim.x + threadIdx.x;
  if (i >= K * N) return;
  int n = i / K, k = i % K;
  out[i] = f2bf(W[(size_t)k * N + n]);
}

// Build embedding A matrix [BL, 192]: A[b*L+l, j*64+c] = x[b,(l+j-1)%L,c] + 1e-10
__global__ void emb_a_k(const float* __restrict__ x, __bf16* __restrict__ out) {
  int i = blockIdx.x * blockDim.x + threadIdx.x;
  if (i >= BLTOK * 192) return;
  int kk = i % 192, bl = i / 192;
  int b = bl >> 9, l = bl & 511;
  int j = kk >> 6, c = kk & 63;
  int l2 = (l + j - 1 + LL) & (LL - 1);
  out[i] = f2bf(x[((size_t)(b * LL + l2)) * CIN + c] + 1e-10f);
}

// Build transposed embedding weight [512, 192]: out[n, j*64+c] = emb_w[j, c, n]
__global__ void emb_w_k(const float* __restrict__ ew, __bf16* __restrict__ out) {
  int i = blockIdx.x * blockDim.x + threadIdx.x;
  if (i >= DD * 192) return;
  int kk = i % 192, n = i / 192;
  int j = kk >> 6, c = kk & 63;
  out[i] = f2bf(ew[(size_t)(j * CIN + c) * DD + n]);
}

// h[b,l,d] += PE(l,d)
__global__ void add_pe_k(float* __restrict__ h) {
  int i = blockIdx.x * blockDim.x + threadIdx.x;
  if (i >= BLTOK * DD) return;
  int d = i & 511;
  int l = (i >> 9) & 511;
  float e  = __expf(-(float)(d & ~1) * 0.017988946f);  // ln(10000)/512
  float ar = (float)l * e;
  h[i] += (d & 1) ? __cosf(ar) : __sinf(ar);
}

// mc[b,tau] = (1/D) * sum_m S[b, (m+tau)%L, m]
__global__ void meancorr_k(const float* __restrict__ S, float* __restrict__ mc) {
  int tau = blockIdx.x, b = blockIdx.y, t = threadIdx.x;
  const float* Sb = S + (size_t)b * LL * LL;
  float s = 0.0f;
  for (int m = t; m < LL; m += 256)
    s += Sb[(size_t)((m + tau) & (LL - 1)) * LL + m];
  __shared__ float red[256];
  red[t] = s; __syncthreads();
  for (int st = 128; st; st >>= 1) { if (t < st) red[t] += red[t + st]; __syncthreads(); }
  if (t == 0) mc[b * LL + tau] = red[0] * (1.0f / (float)DD);
}

// top-6 + softmax per batch
__global__ void topk_k(const float* __restrict__ mc, float* __restrict__ wts, int* __restrict__ dly) {
  __shared__ float sv[512];
  __shared__ float rv[512];
  __shared__ int   ri[512];
  __shared__ float topv[TOPK];
  __shared__ int   topi[TOPK];
  int b = blockIdx.x, t = threadIdx.x;
  sv[t] = mc[b * LL + t];
  __syncthreads();
  for (int k = 0; k < TOPK; ++k) {
    rv[t] = sv[t]; ri[t] = t; __syncthreads();
    for (int st = 256; st; st >>= 1) {
      if (t < st && rv[t + st] > rv[t]) { rv[t] = rv[t + st]; ri[t] = ri[t + st]; }
      __syncthreads();
    }
    if (t == 0) { topv[k] = rv[0]; topi[k] = ri[0]; sv[ri[0]] = -3.0e38f; }
    __syncthreads();
  }
  if (t == 0) {
    float mx = topv[0], se = 0.0f, e[TOPK];
    for (int k = 0; k < TOPK; ++k) { e[k] = __expf(topv[k] - mx); se += e[k]; }
    for (int k = 0; k < TOPK; ++k) { wts[b * TOPK + k] = e[k] / se; dly[b * TOPK + k] = topi[k]; }
  }
}

// out[b,l,d] = sum_k w[b,k] * V[b,(l+delay_k)%L,d]
__global__ void aggregate_k(const float* __restrict__ V, const float* __restrict__ wts,
                            const int* __restrict__ dly, float* __restrict__ out) {
  int i = blockIdx.x * blockDim.x + threadIdx.x;
  if (i >= BLTOK * DD) return;
  int b = i >> 18;
  int l = (i >> 9) & 511;
  int d = i & 511;
  const float* wb = wts + b * TOPK;
  const int*   db = dly + b * TOPK;
  float s = 0.0f;
#pragma unroll
  for (int k = 0; k < TOPK; ++k) {
    int l2 = (l + db[k]) & (LL - 1);
    s += wb[k] * V[((size_t)(b * LL + l2)) * DD + d];
  }
  out[i] = s;
}

// series_decomp: out = x - moving_average(x, 25, replicate pad), per (b,d) sliding window
__global__ void decomp_k(const float* __restrict__ x, float* __restrict__ out) {
  int i = blockIdx.x * blockDim.x + threadIdx.x;
  if (i >= BB * DD) return;
  int b = i >> 9, d = i & 511;
  const float* xb = x + ((size_t)b * LL) * DD + d;
  float* ob = out + ((size_t)b * LL) * DD + d;
  float sum = 13.0f * xb[0];
  for (int j = 1; j <= 12; ++j) sum += xb[(size_t)j * DD];
  for (int l = 0; l < LL; ++l) {
    ob[(size_t)l * DD] = xb[(size_t)l * DD] - sum * (1.0f / 25.0f);
    int hi = l + 13 > LL - 1 ? LL - 1 : l + 13;
    int lo = l - 12 < 0 ? 0 : l - 12;
    sum += xb[(size_t)hi * DD] - xb[(size_t)lo * DD];
  }
}

// layernorm over d (population var), affine
__global__ void ln1_k(const float* __restrict__ x, const float* __restrict__ w,
                      const float* __restrict__ bia, float* __restrict__ o) {
  int bl = blockIdx.x, t = threadIdx.x;
  const float* xr = x + (size_t)bl * DD;
  float s = 0.0f, s2 = 0.0f;
  for (int d = t; d < DD; d += 256) { float v = xr[d]; s += v; s2 += v * v; }
  __shared__ float r1[256], r2[256];
  r1[t] = s; r2[t] = s2; __syncthreads();
  for (int st = 128; st; st >>= 1) { if (t < st) { r1[t] += r1[t + st]; r2[t] += r2[t + st]; } __syncthreads(); }
  float mu = r1[0] * (1.0f / DD);
  float var = r2[0] * (1.0f / DD) - mu * mu;
  float rs = rsqrtf(var + 1e-5f);
  for (int d = t; d < DD; d += 256)
    o[(size_t)bl * DD + d] = (xr[d] - mu) * rs * w[d] + bia[d];
}

// out[b,d] = mean_l( xh - mean_l(xh) )   (reference's my_Layernorm tail + final mean)
__global__ void ln2_k(const float* __restrict__ xh, float* __restrict__ out) {
  int i = blockIdx.x * blockDim.x + threadIdx.x;
  if (i >= BB * DD) return;
  int b = i >> 9, d = i & 511;
  const float* xc = xh + (size_t)b * LL * DD + d;
  float m = 0.0f;
  for (int l = 0; l < LL; ++l) m += xc[(size_t)l * DD];
  m *= (1.0f / LL);
  float s = 0.0f;
  for (int l = 0; l < LL; ++l) s += xc[(size_t)l * DD] - m;
  out[i] = s * (1.0f / LL);
}

// ---------------------------------------------------------------------------
// Host orchestration
// ---------------------------------------------------------------------------
static void gemm_launch(hipStream_t st, const __bf16* A, const __bf16* B,
                        float* C, __bf16* Cbf, const float* bias, const float* resid,
                        int M, int N, int K, long long sA, long long sB, long long sC,
                        int nb, int relu) {
  dim3 g(N / 128, M / 128, nb), blk(256);
  gemm_bf16_k<<<g, blk, 0, st>>>(A, B, C, Cbf, bias, resid, M, N, K, sA, sB, sC, relu);
}

#define EW(n) <<<((n) + 255) / 256, 256, 0, stream>>>

extern "C" void kernel_launch(void* const* d_in, const int* in_sizes, int n_in,
                              void* d_out, int out_size, void* d_ws, size_t ws_size,
                              hipStream_t stream)
{
  (void)in_sizes; (void)n_in; (void)out_size; (void)ws_size;
  const float* x_enc = (const float*)d_in[0];
  const float* emb_w = (const float*)d_in[1];
  const float* Wq = (const float*)d_in[2];
  const float* bq = (const float*)d_in[3];
  const float* Wk = (const float*)d_in[4];
  const float* bk = (const float*)d_in[5];
  const float* Wv = (const float*)d_in[6];
  const float* bv = (const float*)d_in[7];
  const float* Wo = (const float*)d_in[8];
  const float* bo = (const float*)d_in[9];
  const float* c1 = (const float*)d_in[10];
  const float* c2 = (const float*)d_in[11];
  const float* nw = (const float*)d_in[12];
  const float* nb_ = (const float*)d_in[13];
  float* out = (float*)d_out;

  char* base = (char*)d_ws;
  size_t off = 0;
  auto carve = [&](size_t bytes) -> char* {
    char* p = base + off;
    off = (off + bytes + 255) & ~(size_t)255;
    return p;
  };

  const size_t NTOK = (size_t)BLTOK * DD;            // 8.4M elements
  float*  h    = (float*)carve(NTOK * 4);
  float*  vbuf = (float*)carve(NTOK * 4);
  float*  sbuf = (float*)carve(NTOK * 4);            // S (32x512x512) / agg
  float*  x1   = (float*)carve(NTOK * 4);
  __bf16* bfA  = (__bf16*)carve(NTOK * 2);
  __bf16* bfQ  = (__bf16*)carve(NTOK * 2);
  __bf16* bfK  = (__bf16*)carve(NTOK * 2);
  __bf16* bfY1 = (__bf16*)carve((size_t)BLTOK * DFFC * 2);
  __bf16* bfW  = (__bf16*)carve((size_t)DFFC * DD * 2);
  float*  mc   = (float*)carve((size_t)BB * LL * 4);
  float*  wts  = (float*)carve((size_t)BB * 8 * 4);
  int*    dly  = (int*)carve((size_t)BB * 8 * 4);

  // ---- Embedding: circular conv1d(k=3) as GEMM [16384,192]x[192->512] + PE ----
  emb_a_k EW(BLTOK * 192)(x_enc, bfA);
  emb_w_k EW(DD * 192)(emb_w, bfW);
  gemm_launch(stream, bfA, bfW, h, nullptr, nullptr, nullptr,
              BLTOK, DD, 192, 0, 0, 0, 1, 0);
  add_pe_k EW(BLTOK * DD)(h);

  // ---- Encoder layers ----
  for (int layer = 0; layer < 2; ++layer) {
    const float* Wq_l = Wq + (size_t)layer * DD * DD;
    const float* Wk_l = Wk + (size_t)layer * DD * DD;
    const float* Wv_l = Wv + (size_t)layer * DD * DD;
    const float* Wo_l = Wo + (size_t)layer * DD * DD;
    const float* c1_l = c1 + (size_t)layer * DD * DFFC;
    const float* c2_l = c2 + (size_t)layer * DFFC * DD;
    const float* bq_l = bq + (size_t)layer * DD;
    const float* bk_l = bk + (size_t)layer * DD;
    const float* bv_l = bv + (size_t)layer * DD;
    const float* bo_l = bo + (size_t)layer * DD;

    cvt_f32_bf16_k EW(BLTOK * DD)(h, bfA, BLTOK * DD);

    // Q, K directly as bf16; V as fp32 (used by the delay-gather)
    wtrans_k EW(DD * DD)(Wq_l, bfW, DD, DD);
    gemm_launch(stream, bfA, bfW, nullptr, bfQ, bq_l, nullptr, BLTOK, DD, DD, 0, 0, 0, 1, 0);
    wtrans_k EW(DD * DD)(Wk_l, bfW, DD, DD);
    gemm_launch(stream, bfA, bfW, nullptr, bfK, bk_l, nullptr, BLTOK, DD, DD, 0, 0, 0, 1, 0);
    wtrans_k EW(DD * DD)(Wv_l, bfW, DD, DD);
    gemm_launch(stream, bfA, bfW, vbuf, nullptr, bv_l, nullptr, BLTOK, DD, DD, 0, 0, 0, 1, 0);

    // S = Q K^T per batch (replaces FFT autocorrelation; only diag-sums are needed)
    gemm_launch(stream, bfQ, bfK, sbuf, nullptr, nullptr, nullptr,
                LL, LL, DD, (long long)LL * DD, (long long)LL * DD,
                (long long)LL * LL, BB, 0);
    { dim3 g(LL, BB); meancorr_k<<<g, 256, 0, stream>>>(sbuf, mc); }
    topk_k<<<BB, 512, 0, stream>>>(mc, wts, dly);
    aggregate_k EW(BLTOK * DD)(vbuf, wts, dly, sbuf);      // agg -> sbuf

    // O-projection + bias + residual(h)
    cvt_f32_bf16_k EW(BLTOK * DD)(sbuf, bfA, BLTOK * DD);
    wtrans_k EW(DD * DD)(Wo_l, bfW, DD, DD);
    gemm_launch(stream, bfA, bfW, x1, nullptr, bo_l, h, BLTOK, DD, DD, 0, 0, 0, 1, 0);

    // series_decomp 1
    decomp_k EW(BB * DD)(x1, vbuf);                        // x2 -> vbuf

    // FFN: y1 = relu(x2 @ c1w) stored bf16; y2 = y1 @ c2w + x2; decomp -> h
    cvt_f32_bf16_k EW(BLTOK * DD)(vbuf, bfA, BLTOK * DD);
    wtrans_k EW(DD * DFFC)(c1_l, bfW, DD, DFFC);
    gemm_launch(stream, bfA, bfW, nullptr, bfY1, nullptr, nullptr,
                BLTOK, DFFC, DD, 0, 0, 0, 1, 1);
    wtrans_k EW(DFFC * DD)(c2_l, bfW, DFFC, DD);
    gemm_launch(stream, bfY1, bfW, x1, nullptr, nullptr, vbuf,
                BLTOK, DD, DFFC, 0, 0, 0, 1, 0);
    decomp_k EW(BB * DD)(x1, h);
  }

  // ---- final my_Layernorm + time-mean ----
  ln1_k<<<BLTOK, 256, 0, stream>>>(h, nw, nb_, sbuf);
  ln2_k EW(BB * DD)(sbuf, out);
}
